// IterBlock_38500086841954
// MI455X (gfx1250) — matrix operand order repaired and checked by reading, not compiled, not run on hardware
//
#include <hip/hip_runtime.h>

// ---------- CDNA5 vector types ----------
typedef __bf16          v16bf __attribute__((ext_vector_type(16)));
typedef float           v8f   __attribute__((ext_vector_type(8)));
typedef float           v2f   __attribute__((ext_vector_type(2)));
typedef unsigned short  u16x8 __attribute__((ext_vector_type(8)));
typedef unsigned int    u32x4 __attribute__((ext_vector_type(4)));
typedef int             i32x4 __attribute__((ext_vector_type(4)));
typedef int             i32x8 __attribute__((ext_vector_type(8)));

union BF16Frag { v16bf v; u16x8 u[2]; };

__device__ __forceinline__ unsigned short f2bf(float f) {
    unsigned u = __float_as_uint(f);
    u += 0x7FFFu + ((u >> 16) & 1u);          // round-to-nearest-even
    return (unsigned short)(u >> 16);
}

#define HW 384
#define PIX (384 * 384)
#define NN 4096
#define NB 4
#define NC 12

// =====================================================================
// 1) Centered-DFT matrix: F[k,n] = (-1)^(k+n) * exp(-2pi i k n / 384)/sqrt(384)
// =====================================================================
__global__ void k_build_dft(float* __restrict__ fre, float* __restrict__ fim) {
    int n = threadIdx.x, k = blockIdx.x;
    const float s = 0.05103103630798287f;      // 1/sqrt(384)
    float ang = -2.0f * 3.14159265358979323846f * (float)((k * n) % HW) / (float)HW;
    float sgn = ((k + n) & 1) ? -s : s;
    float sv, cv;
    __sincosf(ang, &sv, &cv);
    fre[k * HW + n] = sgn * cv;
    fim[k * HW + n] = sgn * sv;
}

// =====================================================================
// 2) fp32 -> bf16 elementwise (adj conversion)
// =====================================================================
__global__ void k_f32_to_bf16(const float* __restrict__ src,
                              unsigned short* __restrict__ dst, long n) {
    long i = (long)blockIdx.x * blockDim.x + threadIdx.x;
    if (i < n) dst[i] = f2bf(src[i]);
}

// =====================================================================
// 3) Patch-extract + W1 projection, stored TRANSPOSED: S_T[c][col][node] (bf16)
// =====================================================================
__global__ void k_proj1(const float* __restrict__ inp, const float* __restrict__ W1,
                        unsigned short* __restrict__ S_T) {
    int n   = blockIdx.x * blockDim.x + threadIdx.x;   // node 0..4095
    int col = blockIdx.y;                              // 0..255
    int c   = blockIdx.z;                              // 0..1
    int b = col >> 6, o = col & 63;
    int ph = n >> 6, pw = n & 63;
    const float* img = inp + ((size_t)(b * 2 + c)) * PIX;
    float acc = 0.f;
#pragma unroll
    for (int f = 0; f < 36; ++f) {
        int h = ph * 6 + f / 6, w = pw * 6 + f % 6;
        acc += img[h * HW + w] * W1[f * 64 + o];
    }
    S_T[((size_t)(c * 256 + col)) * NN + n] = f2bf(acc);
}

// =====================================================================
// 4) W3 projection, stored TRANSPOSED: S3_T[c][col3][node] (bf16)
// =====================================================================
__global__ void k_proj3(const float* __restrict__ G1, const float* __restrict__ W3,
                        unsigned short* __restrict__ S3T) {
    int n   = blockIdx.x * blockDim.x + threadIdx.x;
    int col = blockIdx.y;                              // 0..143
    int c   = blockIdx.z;
    int b = col / 36, o = col % 36;
    const float* g = G1 + ((size_t)c * NN + n) * 256 + b * 64;
    float acc = 0.f;
#pragma unroll
    for (int f = 0; f < 64; ++f) acc += g[f] * W3[f * 36 + o];
    S3T[((size_t)(c * 144 + col)) * NN + n] = f2bf(acc);
}

// =====================================================================
// 5) bf16 WMMA GEMM with TDM-staged B panel.
//    D[c] (4096 x ncols) = adj_bf16[c] (4096x4096) @ B_T[c] (ncols x 4096, K-major)
//    Block = 4 waves covering 64 M-rows; shared B chunk (NT*16 cols x 64 K)
//    is DMA'd into LDS by the Tensor Data Mover, consumed via ds_load_b128.
// =====================================================================
template <int NT>
__global__ void __launch_bounds__(128)
k_gemm_bf16_wmma(const unsigned short* __restrict__ A,
                 const unsigned short* __restrict__ B,
                 float* __restrict__ D, int K, int ncols, int relu) {
    // LDS B tile: [NT*16 rows(cols of B)][64 K-halves]   (assumed at LDS offset 0)
    __shared__ unsigned short btile[NT * 16 * 64];

    const int tid  = threadIdx.x;
    const int lane = tid & 31;
    const int wave = tid >> 5;                 // 0..3
    const int c    = blockIdx.z;
    const int m0   = blockIdx.y * 64 + wave * 16;
    const int n0   = blockIdx.x * (16 * NT);

    const unsigned short* Ab = A + (size_t)c * K * K;
    const unsigned short* Bb = B + (size_t)c * ncols * K;
    float*                Db = D + (size_t)c * K * ncols;

    v8f acc[NT];
#pragma unroll
    for (int t = 0; t < NT; ++t)
        acc[t] = v8f{0.f, 0.f, 0.f, 0.f, 0.f, 0.f, 0.f, 0.f};

    // A fragment (16x32 bf16): lane holds row m, halves [koff..koff+7],[16+koff..]
    const int mrow = m0 + (lane & 15);
    const int koff = (lane >> 4) * 8;
    const unsigned short* arow = Ab + (size_t)mrow * K + koff;
    const int boff = (lane >> 4) * 8;          // B frag K offset within 32-chunk

    // Constant part of the tensor-DMA descriptor (D# group 1):
    //   data_size=2B, tensor_dim0=K, tensor_dim1=ncols, tile=(64, NT*16),
    //   tensor_dim0_stride=K   (bit packing per CDNA5 ISA 8.4)
    const i32x8 g1 = {
        0x00010000,                                     // data_size=1 (2 bytes)
        (int)((K & 0xFFFF) << 16),                      // tensor_dim0[15:0]
        (int)(((unsigned)K >> 16) | ((ncols & 0xFFFF) << 16)), // dim0 hi | dim1 lo
        (int)(((unsigned)ncols >> 16) | (64u << 16)),   // dim1 hi | tile_dim0=64
        (int)(NT * 16),                                 // tile_dim1
        (int)K,                                         // dim0_stride[31:0]
        0, 0                                            // stride hi / dim1_stride
    };
    const i32x4 gz4 = {0, 0, 0, 0};
    const i32x8 gz8 = {0, 0, 0, 0, 0, 0, 0, 0};

    const unsigned long long bbase =
        (unsigned long long)(size_t)(Bb + (size_t)n0 * K);

    for (int k0 = 0; k0 < K; k0 += 64) {
        if (tid < 32) {
            unsigned long long ga = bbase + (unsigned long long)k0 * 2ull;
            u32x4 g0 = { 1u,                                  // count=1 (valid D#)
                         0u,                                  // lds_addr = 0 (btile)
                         (unsigned)(ga & 0xFFFFFFFFull),
                         (unsigned)((ga >> 32) & 0x1FFFFFFull) | (2u << 30) }; // type=2
            __builtin_amdgcn_tensor_load_to_lds(g0, g1, gz4, gz4, gz8, 0);
            __builtin_amdgcn_s_wait_tensorcnt(0);
        }
        __syncthreads();

#pragma unroll
        for (int kk = 0; kk < 64; kk += 32) {
            BF16Frag a;
            a.u[0] = *(const u16x8*)(arow + k0 + kk);
            a.u[1] = *(const u16x8*)(arow + k0 + kk + 16);
            __builtin_prefetch(arow + k0 + 256, 0, 0);      // global_prefetch_b8
#pragma unroll
            for (int t = 0; t < NT; ++t) {
                const unsigned short* bp =
                    &btile[(t * 16 + (lane & 15)) * 64 + kk + boff];
                BF16Frag b;
                b.u[0] = *(const u16x8*)(bp);
                b.u[1] = *(const u16x8*)(bp + 16);
                acc[t] = __builtin_amdgcn_wmma_f32_16x16x32_bf16(
                    false, a.v, false, b.v, (short)0, acc[t], false, false);
            }
        }
        __syncthreads();
    }

#pragma unroll
    for (int t = 0; t < NT; ++t) {
        const int col = n0 + t * 16 + (lane & 15);
#pragma unroll
        for (int v = 0; v < 8; ++v) {
            int row = m0 + v + ((lane >> 4) ? 8 : 0);
            float val = acc[t][v];
            if (relu) val = fmaxf(val, 0.f);
            Db[(size_t)row * ncols + col] = val;
        }
    }
}

// =====================================================================
// 6) Complex f32 WMMA GEMM (16x16x4) with LDS-staged B panel.
//    Block = 4 waves, 64 M-rows x 16 N-cols; whole 384x16 complex B panel
//    staged in LDS (conjugation folded into the staging store).
// =====================================================================
__global__ void __launch_bounds__(128)
k_cgemm_f32_wmma(const float* __restrict__ Ar, const float* __restrict__ Ai, long sA,
                 const float* __restrict__ Br, const float* __restrict__ Bi, long sB,
                 float* __restrict__ Dr, float* __restrict__ Di, long sD,
                 int N, int K, float aim_sign, float bim_sign) {
    __shared__ float sbr[HW * 16];
    __shared__ float sbi[HW * 16];

    const int tid  = threadIdx.x;
    const int lane = tid & 31;
    const int wave = tid >> 5;
    const long img = blockIdx.z;

    const float* ar_ = Ar + img * sA;
    const float* ai_ = Ai + img * sA;
    const float* br_ = Br + img * sB;
    const float* bi_ = Bi + img * sB;
    float* dr_ = Dr + img * sD;
    float* di_ = Di + img * sD;

    const int m0   = blockIdx.y * 64 + wave * 16;
    const int n0   = blockIdx.x * 16;
    const int mrow = m0 + (lane & 15);
    const int ncol = lane & 15;
    const int ka   = (lane >> 4) * 2;

    // cooperative coalesced staging of the B panel (16 floats = 64B per row)
    for (int r = tid >> 4; r < K; r += 8) {
        int cc = tid & 15;
        sbr[r * 16 + cc] = br_[(size_t)r * N + n0 + cc];
        sbi[r * 16 + cc] = bi_[(size_t)r * N + n0 + cc] * bim_sign;
    }
    __syncthreads();

    v8f accr = v8f{0.f, 0.f, 0.f, 0.f, 0.f, 0.f, 0.f, 0.f};
    v8f acci = v8f{0.f, 0.f, 0.f, 0.f, 0.f, 0.f, 0.f, 0.f};

    for (int k0 = 0; k0 < K; k0 += 4) {
        v2f arf = *(const v2f*)(ar_ + (size_t)mrow * K + k0 + ka);
        v2f aif = *(const v2f*)(ai_ + (size_t)mrow * K + k0 + ka);
        aif *= aim_sign;
        int kb = k0 + ka;
        v2f brf, bif;
        brf.x = sbr[(kb)     * 16 + ncol];
        brf.y = sbr[(kb + 1) * 16 + ncol];
        bif.x = sbi[(kb)     * 16 + ncol];
        bif.y = sbi[(kb + 1) * 16 + ncol];
        v2f nai = -aif;
        // Dr += Ar*Br - Ai*Bi ; Di += Ar*Bi + Ai*Br (f32 WMMA has no A/B NEG)
        accr = __builtin_amdgcn_wmma_f32_16x16x4_f32(false, arf, false, brf, (short)0, accr, false, false);
        accr = __builtin_amdgcn_wmma_f32_16x16x4_f32(false, nai, false, bif, (short)0, accr, false, false);
        acci = __builtin_amdgcn_wmma_f32_16x16x4_f32(false, arf, false, bif, (short)0, acci, false, false);
        acci = __builtin_amdgcn_wmma_f32_16x16x4_f32(false, aif, false, brf, (short)0, acci, false, false);
    }

    const int nc = blockIdx.x * 16 + ncol;
#pragma unroll
    for (int v = 0; v < 8; ++v) {
        int row = m0 + v + ((lane >> 4) ? 8 : 0);
        dr_[(size_t)row * N + nc] = accr[v];
        di_[(size_t)row * N + nc] = acci[v];
    }
}

// =====================================================================
// 7) Direct 3x3 SAME conv, one batch image, optional relu
// =====================================================================
__global__ void k_conv3x3(const float* __restrict__ in, const float* __restrict__ wgt,
                          const float* __restrict__ bias, float* __restrict__ out,
                          int Cin, int relu) {
    int x  = threadIdx.x;       // 0..383
    int y  = blockIdx.x;        // 0..383
    int co = blockIdx.y;
    float acc = bias[co];
    for (int ci = 0; ci < Cin; ++ci) {
        const float* ip = in + (size_t)ci * PIX;
        const float* wp = wgt + ((size_t)co * Cin + ci) * 9;
#pragma unroll
        for (int ky = 0; ky < 3; ++ky) {
            int yy = y + ky - 1;
            if (yy < 0 || yy >= HW) continue;
#pragma unroll
            for (int kx = 0; kx < 3; ++kx) {
                int xx = x + kx - 1;
                if (xx < 0 || xx >= HW) continue;
                acc += ip[yy * HW + xx] * wp[ky * 3 + kx];
            }
        }
    }
    if (relu) acc = fmaxf(acc, 0.f);
    out[(size_t)co * PIX + y * HW + x] = acc;
}

// =====================================================================
// 8) out = input + (1-arf)*tmp2 + arf*feature2image(T3)
// =====================================================================
__global__ void k_combine(const float* __restrict__ inp, const float* __restrict__ tmp2,
                          const float* __restrict__ T3, const float* __restrict__ arf,
                          float* __restrict__ outp) {
    size_t i = (size_t)blockIdx.x * blockDim.x + threadIdx.x;
    if (i >= (size_t)NB * 2 * PIX) return;
    int w  = i % HW;
    int h  = (i / HW) % HW;
    int ch = (i / PIX) % 2;
    int b  = (int)(i / ((size_t)2 * PIX));
    int n = (h / 6) * 64 + (w / 6);
    int f = (h % 6) * 6 + (w % 6);
    float t3 = T3[((size_t)ch * NN + n) * 144 + b * 36 + f];
    float a  = arf[0];
    outp[i] = inp[i] + (1.f - a) * tmp2[i] + a * t3;
}

// =====================================================================
// 9) coil expansion: (out0 + i out1) * csm, laid out [C][B][H][W]
// =====================================================================
__global__ void k_make_coil(const float* __restrict__ outp, const float* __restrict__ csr,
                            const float* __restrict__ csi, float* __restrict__ kr,
                            float* __restrict__ ki) {
    size_t i = (size_t)blockIdx.x * blockDim.x + threadIdx.x;
    if (i >= (size_t)NC * NB * PIX) return;
    int pix = i % PIX;
    int b   = (int)((i / PIX) % NB);
    int c   = (int)(i / ((size_t)NB * PIX));
    float xr = outp[((size_t)b * 2 + 0) * PIX + pix];
    float xi = outp[((size_t)b * 2 + 1) * PIX + pix];
    float cr = csr[((size_t)b * NC + c) * PIX + pix];
    float cm = csi[((size_t)b * NC + c) * PIX + pix];
    kr[i] = xr * cr - xi * cm;
    ki[i] = xr * cm + xi * cr;
}

// =====================================================================
// 10) k-space data consistency (in place)
// =====================================================================
__global__ void k_dc(float* __restrict__ kr, float* __restrict__ ki,
                     const int* __restrict__ mask, const float* __restrict__ tkr,
                     const float* __restrict__ tki) {
    size_t i = (size_t)blockIdx.x * blockDim.x + threadIdx.x;
    if (i >= (size_t)NC * NB * PIX) return;
    int pix = i % PIX;
    float m  = (float)mask[pix];
    float om = 1.f - m;
    kr[i] = om * kr[i] + 0.999999f * m * tkr[i];
    ki[i] = om * ki[i] + 0.999999f * m * tki[i];
}

// =====================================================================
// 11) coil combine: res = sum_c oi * conj(csm) -> d_out [B,2,H,W]
// =====================================================================
__global__ void k_coil_combine(const float* __restrict__ oir, const float* __restrict__ oii,
                               const float* __restrict__ csr, const float* __restrict__ csi,
                               float* __restrict__ out) {
    size_t i = (size_t)blockIdx.x * blockDim.x + threadIdx.x;
    if (i >= (size_t)NB * PIX) return;
    int pix = i % PIX;
    int b   = (int)(i / PIX);
    float rr = 0.f, ri = 0.f;
    for (int c = 0; c < NC; ++c) {
        size_t idx = ((size_t)c * NB + b) * PIX + pix;
        float cr = csr[((size_t)b * NC + c) * PIX + pix];
        float cm = csi[((size_t)b * NC + c) * PIX + pix];
        float yr = oir[idx], yi = oii[idx];
        rr += yr * cr + yi * cm;
        ri += yi * cr - yr * cm;
    }
    out[((size_t)b * 2 + 0) * PIX + pix] = rr;
    out[((size_t)b * 2 + 1) * PIX + pix] = ri;
}

// =====================================================================
// host-side launcher
// =====================================================================
extern "C" void kernel_launch(void* const* d_in, const int* in_sizes, int n_in,
                              void* d_out, int out_size, void* d_ws, size_t ws_size,
                              hipStream_t stream) {
    const float* input_data = (const float*)d_in[0];
    const float* adj        = (const float*)d_in[1];
    const int*   mask       = (const int*)d_in[2];
    const float* csr        = (const float*)d_in[3];
    const float* csi        = (const float*)d_in[4];
    const float* tkr        = (const float*)d_in[5];
    const float* tki        = (const float*)d_in[6];
    const float* W1         = (const float*)d_in[7];
    const float* W3         = (const float*)d_in[8];
    const float* arf        = (const float*)d_in[9];
    const float* cw1        = (const float*)d_in[10];
    const float* cb1        = (const float*)d_in[11];
    const float* cw2        = (const float*)d_in[12];
    const float* cb2        = (const float*)d_in[13];
    const float* cw3        = (const float*)d_in[14];
    const float* cb3        = (const float*)d_in[15];
    float* out = (float*)d_out;

    // ---- workspace carving (bytes, 256-aligned) ----
    char* ws = (char*)d_ws;
    size_t off = 0;
    auto carve = [&](size_t bytes) -> void* {
        void* p = ws + off;
        off = (off + bytes + 255) & ~(size_t)255;
        return p;
    };
    float*          FRE  = (float*)carve((size_t)PIX * 4);
    float*          FIM  = (float*)carve((size_t)PIX * 4);
    unsigned short* ADJB = (unsigned short*)carve((size_t)2 * NN * NN * 2);
    unsigned short* S_T  = (unsigned short*)carve((size_t)2 * 256 * NN * 2);
    float*          G1   = (float*)carve((size_t)2 * NN * 256 * 4);
    unsigned short* S3T  = (unsigned short*)carve((size_t)2 * 144 * NN * 2);
    float*          T3   = (float*)carve((size_t)2 * NN * 144 * 4);
    float*          H1   = (float*)carve((size_t)64 * PIX * 4);
    float*          H2   = (float*)carve((size_t)64 * PIX * 4);
    float*          TMP2 = (float*)carve((size_t)NB * 2 * PIX * 4);
    float*          OUTP = (float*)carve((size_t)NB * 2 * PIX * 4);
    float*          KAR  = (float*)carve((size_t)NC * NB * PIX * 4);
    float*          KAI  = (float*)carve((size_t)NC * NB * PIX * 4);
    float*          KBR  = (float*)carve((size_t)NC * NB * PIX * 4);
    float*          KBI  = (float*)carve((size_t)NC * NB * PIX * 4);

    // ---- DFT matrix + bf16 adj ----
    k_build_dft<<<dim3(HW), dim3(HW), 0, stream>>>(FRE, FIM);
    {
        long n = (long)2 * NN * NN;
        k_f32_to_bf16<<<dim3((unsigned)((n + 255) / 256)), dim3(256), 0, stream>>>(adj, ADJB, n);
    }

    // ---- GCN branch (WMMA bf16 + TDM staging) ----
    k_proj1<<<dim3(NN / 256, 256, 2), dim3(256), 0, stream>>>(input_data, W1, S_T);
    k_gemm_bf16_wmma<4><<<dim3(256 / 64, NN / 64, 2), dim3(128), 0, stream>>>(ADJB, S_T, G1, NN, 256, 1);
    k_proj3<<<dim3(NN / 256, 144, 2), dim3(256), 0, stream>>>(G1, W3, S3T);
    k_gemm_bf16_wmma<3><<<dim3(144 / 48, NN / 64, 2), dim3(128), 0, stream>>>(ADJB, S3T, T3, NN, 144, 0);

    // ---- CNN branch (per-batch, hidden buffers reused) ----
    for (int b = 0; b < NB; ++b) {
        const float* in_b = input_data + (size_t)b * 2 * PIX;
        k_conv3x3<<<dim3(HW, 64), dim3(HW), 0, stream>>>(in_b, cw1, cb1, H1, 2, 1);
        k_conv3x3<<<dim3(HW, 64), dim3(HW), 0, stream>>>(H1, cw2, cb2, H2, 64, 1);
        k_conv3x3<<<dim3(HW, 2), dim3(HW), 0, stream>>>(H2, cw3, cb3,
                                                        TMP2 + (size_t)b * 2 * PIX, 64, 0);
    }

    // ---- combine ----
    {
        size_t n = (size_t)NB * 2 * PIX;
        k_combine<<<dim3((unsigned)((n + 255) / 256)), dim3(256), 0, stream>>>(
            input_data, TMP2, T3, arf, OUTP);
    }

    // ---- coil expansion ----
    {
        size_t n = (size_t)NC * NB * PIX;
        k_make_coil<<<dim3((unsigned)((n + 255) / 256)), dim3(256), 0, stream>>>(
            OUTP, csr, csi, KAR, KAI);
    }

    const long simg = (long)PIX;
    const dim3 cg(HW / 16, HW / 64, NC * NB);

    // fft2c:  Z = X @ F  (F symmetric), then Y = F @ Z
    k_cgemm_f32_wmma<<<cg, dim3(128), 0, stream>>>(KAR, KAI, simg, FRE, FIM, 0,
                                                   KBR, KBI, simg, HW, HW, 1.f, 1.f);
    k_cgemm_f32_wmma<<<cg, dim3(128), 0, stream>>>(FRE, FIM, 0, KBR, KBI, simg,
                                                   KAR, KAI, simg, HW, HW, 1.f, 1.f);

    // ---- data consistency ----
    {
        size_t n = (size_t)NC * NB * PIX;
        k_dc<<<dim3((unsigned)((n + 255) / 256)), dim3(256), 0, stream>>>(
            KAR, KAI, mask, tkr, tki);
    }

    // ifft2c: conj(F) on both sides
    k_cgemm_f32_wmma<<<cg, dim3(128), 0, stream>>>(KAR, KAI, simg, FRE, FIM, 0,
                                                   KBR, KBI, simg, HW, HW, 1.f, -1.f);
    k_cgemm_f32_wmma<<<cg, dim3(128), 0, stream>>>(FRE, FIM, 0, KBR, KBI, simg,
                                                   KAR, KAI, simg, HW, HW, -1.f, 1.f);

    // ---- coil combine -> output ----
    {
        size_t n = (size_t)NB * PIX;
        k_coil_combine<<<dim3((unsigned)((n + 255) / 256)), dim3(256), 0, stream>>>(
            KAR, KAI, csr, csi, out);
    }
}
